// UnsupNet_31009663877308
// MI455X (gfx1250) — compile-verified, matrix operand order checked by reading, not run on hardware
//
#include <hip/hip_runtime.h>
#include <hip/hip_bf16.h>

typedef __attribute__((ext_vector_type(16))) __bf16 v16bf;
typedef __attribute__((ext_vector_type(8)))  __bf16 v8bf;
typedef __attribute__((ext_vector_type(8)))  float  v8f;

#define NB 64   // batch
#define IMG 64  // image H=W

// ---------------------------------------------------------------------------
// Pack weights into WMMA B-fragment order (bf16):
//   dst[((nt*KC + kc)*32 + lane)*16 + slot] = W[n = nt*16 + lane&15,
//        k = kc*32 + (slot>=8?16:0) + (lane>>4)*8 + (slot&7)]   (0 if k>=K)
// src element (n,k) fetched at src[n*sN + k*sK] (covers OIHW and KxN layouts).
// ---------------------------------------------------------------------------
__global__ void pack_wmma_b(const float* __restrict__ src, __bf16* __restrict__ dst,
                            int K, int KC, int NTILES, int sN, int sK) {
  int idx = blockIdx.x * blockDim.x + threadIdx.x;
  int total = NTILES * KC * 512;
  if (idx >= total) return;
  int slot = idx & 15;
  int lane = (idx >> 4) & 31;
  int rest = idx >> 9;            // nt*KC + kc
  int kc = rest % KC;
  int nt = rest / KC;
  int k = kc * 32 + ((slot >= 8) ? 16 : 0) + (lane >> 4) * 8 + (slot & 7);
  int n = nt * 16 + (lane & 15);
  float v = (k < K) ? src[(size_t)n * sN + (size_t)k * sK] : 0.0f;
  dst[idx] = (__bf16)v;
}

// ---------------------------------------------------------------------------
// Conv 3x3 (pad 1) implicit GEMM on v_wmma_f32_16x16x32_bf16.
// M = NB*H*W positions, N = COUT, K = CIN*9 (zero-padded to mult of 32).
// Block (32, COUT/16): waves share one LDS A-tile staged in fragment layout;
// B fragments are direct 32B global loads from pre-packed weights.
// ---------------------------------------------------------------------------
template<int CIN, int COUT, int H, int W, bool IN_F32>
__global__ void conv3x3_wmma(const float* __restrict__ inF,
                             const __bf16* __restrict__ inB,
                             const __bf16* __restrict__ wpk,
                             const float* __restrict__ bias,
                             __bf16* __restrict__ out) {
  constexpr int K  = CIN * 9;
  constexpr int KC = (K + 31) / 32;
  constexpr int NW = COUT / 16;
  constexpr int NTHREADS = NW * 32;
  __shared__ v16bf Asf[32];   // A tile in fragment layout: 32 lanes x 16 bf16

  const int lane = threadIdx.x;            // 0..31
  const int wv   = threadIdx.y;            // 0..NW-1  (= n-tile)
  const int tid  = wv * 32 + lane;
  const int mBase = blockIdx.x * 16;
  const int laneHalf = lane >> 4;
  const int r = lane & 15;
  const __bf16* wbase = wpk + ((size_t)wv * KC * 32 + lane) * 16;

  v8f c = {};
  for (int kc = 0; kc < KC; ++kc) {
    // B fragment: direct 32-byte load (2x global_load_b128), overlaps barrier
    v16bf bfrag = *(const v16bf*)(wbase + (size_t)kc * 512);
    __syncthreads();
    // Stage A tile (im2col gather) directly into fragment layout
    for (int idx = tid; idx < 512; idx += NTHREADS) {
      int row = idx >> 5, k = idx & 31;
      int kg = kc * 32 + k;
      float v = 0.0f;
      if (kg < K) {
        int cin = kg / 9, f = kg - cin * 9;
        int fy = f / 3, fx = f - fy * 3;
        int p = mBase + row;
        int bI = p / (H * W);
        int rem = p - bI * (H * W);
        int y = rem / W, x = rem - y * W;
        int iy = y + fy - 1, ix = x + fx - 1;
        if (iy >= 0 && iy < H && ix >= 0 && ix < W) {
          int off = ((bI * CIN + cin) * H + iy) * W + ix;
          if constexpr (IN_F32) v = inF[off];
          else                  v = (float)inB[off];
        }
      }
      int dl = ((k >> 3) & 1) * 16 + row;       // dest lane
      int ds = ((k >= 16) ? 8 : 0) + (k & 7);   // dest slot
      ((__bf16*)&Asf[dl])[ds] = (__bf16)v;
    }
    __syncthreads();
    v16bf a = Asf[lane];                        // 2x ds_load_b128
    c = __builtin_amdgcn_wmma_f32_16x16x32_bf16(false, a, false, bfrag,
                                                (short)0, c, false, false);
  }
  // bias + ReLU + NCHW store (bf16)
#pragma unroll
  for (int v = 0; v < 8; ++v) {
    int m  = mBase + v + 8 * laneHalf;
    int co = wv * 16 + r;
    float val = c[v] + bias[co];
    val = val > 0.0f ? val : 0.0f;
    int bI = m / (H * W);
    int rem = m - bI * (H * W);
    int y = rem / W, x = rem - y * W;
    out[((bI * COUT + co) * H + y) * W + x] = (__bf16)val;
  }
}

// ---------------------------------------------------------------------------
// 2x2 maxpool, stride 2, bf16 NCHW
// ---------------------------------------------------------------------------
__global__ void maxpool2_bf16(const __bf16* __restrict__ in,
                              __bf16* __restrict__ out,
                              int C, int Hin, int Win, int total) {
  int idx = blockIdx.x * blockDim.x + threadIdx.x;
  if (idx >= total) return;
  int Wo = Win >> 1, Ho = Hin >> 1;
  int wo = idx % Wo; int t = idx / Wo;
  int ho = t % Ho;   t /= Ho;
  int cc = t % C;    int b = t / C;
  const __bf16* p = in + ((size_t)(b * C + cc) * Hin + 2 * ho) * Win + 2 * wo;
  float v0 = (float)p[0], v1 = (float)p[1];
  float v2 = (float)p[Win], v3 = (float)p[Win + 1];
  out[idx] = (__bf16)fmaxf(fmaxf(v0, v1), fmaxf(v2, v3));
}

// ---------------------------------------------------------------------------
// FC1: (64 x 8192) @ (8192 x 1024) + b5, sigmoid. Register-only WMMA:
// A fragments are two 16B loads from k-contiguous activations, B fragments
// are 32B loads from packed w5. No LDS, no barriers. grid (4, 64), block 32.
// ---------------------------------------------------------------------------
__global__ void fc1_wmma(const __bf16* __restrict__ f,
                         const __bf16* __restrict__ w5pk,
                         const float* __restrict__ b5,
                         float* __restrict__ out) {
  constexpr int KTOT = 8192, N = 1024, KC = KTOT / 32;
  const int lane = threadIdx.x;
  const int mBase = blockIdx.x * 16;
  const int nt = blockIdx.y;
  const int laneHalf = lane >> 4;
  const int r = lane & 15;
  const __bf16* arow  = f + (size_t)(mBase + r) * KTOT + laneHalf * 8;
  const __bf16* wbase = w5pk + ((size_t)nt * KC * 32 + lane) * 16;

  v8f c = {};
  for (int kc = 0; kc < KC; ++kc) {
    union { v16bf v; v8bf h[2]; } au;
    au.h[0] = *(const v8bf*)(arow + kc * 32);        // k = kc*32 + half*8 + 0..7
    au.h[1] = *(const v8bf*)(arow + kc * 32 + 16);   // k = kc*32+16 + half*8 + 0..7
    v16bf bfrag = *(const v16bf*)(wbase + (size_t)kc * 512);
    c = __builtin_amdgcn_wmma_f32_16x16x32_bf16(false, au.v, false, bfrag,
                                                (short)0, c, false, false);
  }
#pragma unroll
  for (int v = 0; v < 8; ++v) {
    int m = mBase + v + 8 * laneHalf;
    int n = nt * 16 + r;
    float val = c[v] + b5[n];
    out[(size_t)m * N + n] = 1.0f / (1.0f + expf(-val));
  }
}

// ---------------------------------------------------------------------------
// FC2 (1024 -> 8) + softmax; one thread per batch item
// ---------------------------------------------------------------------------
__global__ void fc2_softmax(const float* __restrict__ f,
                            const float* __restrict__ w6,
                            const float* __restrict__ b6,
                            float* __restrict__ hWs,
                            float* __restrict__ hOut) {
  int b = blockIdx.x * blockDim.x + threadIdx.x;
  if (b >= NB) return;
  float acc[8];
#pragma unroll
  for (int n = 0; n < 8; ++n) acc[n] = b6[n];
  for (int k = 0; k < 1024; ++k) {
    float fv = f[b * 1024 + k];
#pragma unroll
    for (int n = 0; n < 8; ++n) acc[n] += fv * w6[k * 8 + n];
  }
  float mx = acc[0];
#pragma unroll
  for (int n = 1; n < 8; ++n) mx = fmaxf(mx, acc[n]);
  float s = 0.0f;
#pragma unroll
  for (int n = 0; n < 8; ++n) { acc[n] = expf(acc[n] - mx); s += acc[n]; }
  float inv = 1.0f / s;
#pragma unroll
  for (int n = 0; n < 8; ++n) {
    hWs[b * 8 + n]  = acc[n] * inv;
    hOut[b * 8 + n] = acc[n] * inv;
  }
}

// ---------------------------------------------------------------------------
// DLT: 8x8 Gauss-Jordan with partial pivoting, then 3x3 inverse (adjugate).
// One thread per batch item; writes Hinv (row-major 9 floats).
// ---------------------------------------------------------------------------
__global__ void dlt_solve(const float* __restrict__ h4,
                          const float* __restrict__ CA,
                          float* __restrict__ HinvOut) {
  int b = blockIdx.x * blockDim.x + threadIdx.x;
  if (b >= NB) return;
  float ca[8], cb[8];
#pragma unroll
  for (int i = 0; i < 8; ++i) {
    ca[i] = CA[b * 8 + i];
    cb[i] = ca[i] + h4[b * 8 + i];
  }
  float Aug[8][9];
#pragma unroll
  for (int i = 0; i < 4; ++i) {
    float U  = ca[2 * i],  V  = ca[2 * i + 1];
    float U_ = cb[2 * i],  V_ = cb[2 * i + 1];
    float* r0 = Aug[2 * i];
    float* r1 = Aug[2 * i + 1];
    r0[0] = 0.f;  r0[1] = 0.f;  r0[2] = 0.f;
    r0[3] = -U;   r0[4] = V;    r0[5] = -1.f;
    r0[6] = V_ * U; r0[7] = V_ * V; r0[8] = -V_;
    r1[0] = U;    r1[1] = V;    r1[2] = 1.f;
    r1[3] = 0.f;  r1[4] = 0.f;  r1[5] = 0.f;
    r1[6] = -U_ * U; r1[7] = -U_ * V; r1[8] = U_;
  }
  for (int col = 0; col < 8; ++col) {
    int piv = col; float best = fabsf(Aug[col][col]);
    for (int rr = col + 1; rr < 8; ++rr) {
      float v = fabsf(Aug[rr][col]);
      if (v > best) { best = v; piv = rr; }
    }
    if (piv != col)
      for (int cc = 0; cc < 9; ++cc) {
        float t = Aug[col][cc]; Aug[col][cc] = Aug[piv][cc]; Aug[piv][cc] = t;
      }
    float inv = 1.0f / Aug[col][col];
    for (int cc = col; cc < 9; ++cc) Aug[col][cc] *= inv;
    for (int rr = 0; rr < 8; ++rr) {
      if (rr == col) continue;
      float fct = Aug[rr][col];
      for (int cc = col; cc < 9; ++cc) Aug[rr][cc] -= fct * Aug[col][cc];
    }
  }
  float a = Aug[0][8], b2 = Aug[1][8], c2 = Aug[2][8];
  float d = Aug[3][8], e  = Aug[4][8], f2 = Aug[5][8];
  float g = Aug[6][8], h2 = Aug[7][8], i2 = 1.0f;
  float det = a * (e * i2 - f2 * h2) - b2 * (d * i2 - f2 * g)
            + c2 * (d * h2 - e * g);
  float idet = 1.0f / det;
  float* Hi = HinvOut + b * 9;
  Hi[0] = (e * i2 - f2 * h2) * idet;
  Hi[1] = (c2 * h2 - b2 * i2) * idet;
  Hi[2] = (b2 * f2 - c2 * e) * idet;
  Hi[3] = (f2 * g - d * i2) * idet;
  Hi[4] = (a * i2 - c2 * g) * idet;
  Hi[5] = (c2 * d - a * f2) * idet;
  Hi[6] = (d * h2 - e * g) * idet;
  Hi[7] = (b2 * g - a * h2) * idet;
  Hi[8] = (a * e - b2 * d) * idet;
}

// ---------------------------------------------------------------------------
// Perspective warp with bilinear sampling, zero outside; 1 thread / pixel
// ---------------------------------------------------------------------------
__global__ void warp_persp(const float* __restrict__ xa,
                           const float* __restrict__ Hinv,
                           float* __restrict__ out) {
  int idx = blockIdx.x * blockDim.x + threadIdx.x;
  if (idx >= NB * IMG * IMG) return;
  int b = idx >> 12;
  int pix = idx & 4095;
  int y = pix >> 6, x = pix & 63;
  const float* Hi = Hinv + b * 9;
  float fx = (float)x, fy = (float)y;
  float sxn = Hi[0] * fx + Hi[1] * fy + Hi[2];
  float syn = Hi[3] * fx + Hi[4] * fy + Hi[5];
  float sw  = Hi[6] * fx + Hi[7] * fy + Hi[8];
  float sx = sxn / sw, sy = syn / sw;
  float x0 = floorf(sx), y0 = floorf(sy);
  float wx = sx - x0,   wy = sy - y0;
  const float* img = xa + (size_t)b * IMG * IMG;
  auto gather = [&](float gx, float gy) -> float {
    bool valid = (gx >= 0.f) && (gx < (float)IMG) && (gy >= 0.f) && (gy < (float)IMG);
    int ix = (int)fminf(fmaxf(gx, 0.f), (float)(IMG - 1));
    int iy = (int)fminf(fmaxf(gy, 0.f), (float)(IMG - 1));
    return valid ? img[iy * IMG + ix] : 0.0f;
  };
  float v00 = gather(x0,       y0);
  float v01 = gather(x0 + 1.f, y0);
  float v10 = gather(x0,       y0 + 1.f);
  float v11 = gather(x0 + 1.f, y0 + 1.f);
  float top = v00 * (1.f - wx) + v01 * wx;
  float bot = v10 * (1.f - wx) + v11 * wx;
  out[idx] = top * (1.f - wy) + bot * wy;
}

// ---------------------------------------------------------------------------
extern "C" void kernel_launch(void* const* d_in, const int* in_sizes, int n_in,
                              void* d_out, int out_size, void* d_ws, size_t ws_size,
                              hipStream_t stream) {
  // inputs: xa, xb, x, C_A, then params dict in insertion order
  const float* xa  = (const float*)d_in[0];
  const float* x   = (const float*)d_in[2];
  const float* CA  = (const float*)d_in[3];
  const float* w1a = (const float*)d_in[4];  const float* b1a = (const float*)d_in[5];
  const float* w1b = (const float*)d_in[6];  const float* b1b = (const float*)d_in[7];
  const float* w2a = (const float*)d_in[8];  const float* b2a = (const float*)d_in[9];
  const float* w2b = (const float*)d_in[10]; const float* b2b = (const float*)d_in[11];
  const float* w3a = (const float*)d_in[12]; const float* b3a = (const float*)d_in[13];
  const float* w3b = (const float*)d_in[14]; const float* b3b = (const float*)d_in[15];
  const float* w4a = (const float*)d_in[16]; const float* b4a = (const float*)d_in[17];
  const float* w4b = (const float*)d_in[18]; const float* b4b = (const float*)d_in[19];
  const float* w5  = (const float*)d_in[20]; const float* b5  = (const float*)d_in[21];
  const float* w6  = (const float*)d_in[22]; const float* b6  = (const float*)d_in[23];

  // ---- workspace layout (bf16 acts, packed bf16 weights, f32 tails) ----
  const size_t ACT = (size_t)NB * 64 * IMG * IMG;   // 16,777,216 elems
  __bf16* actA = (__bf16*)d_ws;
  __bf16* actB = actA + ACT;
  __bf16* wp1a = actB + ACT;        // 4*2*512    = 4096
  __bf16* wp1b = wp1a + 4096;       // 4*18*512   = 36864
  __bf16* wp2a = wp1b + 36864;
  __bf16* wp2b = wp2a + 36864;
  __bf16* wp3a = wp2b + 36864;      // 8*18*512   = 73728
  __bf16* wp3b = wp3a + 73728;      // 8*36*512   = 147456
  __bf16* wp4a = wp3b + 147456;
  __bf16* wp4b = wp4a + 147456;
  __bf16* wp5  = wp4b + 147456;     // 64*256*512 = 8,388,608
  float*  fF   = (float*)(wp5 + 8388608);           // 64*1024
  float*  hWs  = fF + NB * 1024;                    // 64*8
  float*  Hinv = hWs + NB * 8;                      // 64*9

  float* outW = (float*)d_out;            // warped: 262144
  float* outH = outW + NB * IMG * IMG;    // h: 512

  // ---- pack all WMMA B operands to bf16 fragment order ----
  pack_wmma_b<<<(4096   + 255) / 256, 256, 0, stream>>>(w1a, wp1a,   54,  2,  4,   54, 1);
  pack_wmma_b<<<(36864  + 255) / 256, 256, 0, stream>>>(w1b, wp1b,  576, 18,  4,  576, 1);
  pack_wmma_b<<<(36864  + 255) / 256, 256, 0, stream>>>(w2a, wp2a,  576, 18,  4,  576, 1);
  pack_wmma_b<<<(36864  + 255) / 256, 256, 0, stream>>>(w2b, wp2b,  576, 18,  4,  576, 1);
  pack_wmma_b<<<(73728  + 255) / 256, 256, 0, stream>>>(w3a, wp3a,  576, 18,  8,  576, 1);
  pack_wmma_b<<<(147456 + 255) / 256, 256, 0, stream>>>(w3b, wp3b, 1152, 36,  8, 1152, 1);
  pack_wmma_b<<<(147456 + 255) / 256, 256, 0, stream>>>(w4a, wp4a, 1152, 36,  8, 1152, 1);
  pack_wmma_b<<<(147456 + 255) / 256, 256, 0, stream>>>(w4b, wp4b, 1152, 36,  8, 1152, 1);
  pack_wmma_b<<<(8388608 + 255) / 256, 256, 0, stream>>>(w5, wp5, 8192, 256, 64, 1, 1024);

  // ---- conv trunk ----
  conv3x3_wmma<6, 64, 64, 64, true ><<<NB * 64 * 64 / 16, dim3(32, 4), 0, stream>>>(x, nullptr, wp1a, b1a, actA);
  conv3x3_wmma<64, 64, 64, 64, false><<<NB * 64 * 64 / 16, dim3(32, 4), 0, stream>>>(nullptr, actA, wp1b, b1b, actB);
  maxpool2_bf16<<<(NB * 64 * 32 * 32 + 255) / 256, 256, 0, stream>>>(actB, actA, 64, 64, 64, NB * 64 * 32 * 32);
  conv3x3_wmma<64, 64, 32, 32, false><<<NB * 32 * 32 / 16, dim3(32, 4), 0, stream>>>(nullptr, actA, wp2a, b2a, actB);
  conv3x3_wmma<64, 64, 32, 32, false><<<NB * 32 * 32 / 16, dim3(32, 4), 0, stream>>>(nullptr, actB, wp2b, b2b, actA);
  maxpool2_bf16<<<(NB * 64 * 16 * 16 + 255) / 256, 256, 0, stream>>>(actA, actB, 64, 32, 32, NB * 64 * 16 * 16);
  conv3x3_wmma<64, 128, 16, 16, false><<<NB * 16 * 16 / 16, dim3(32, 8), 0, stream>>>(nullptr, actB, wp3a, b3a, actA);
  conv3x3_wmma<128, 128, 16, 16, false><<<NB * 16 * 16 / 16, dim3(32, 8), 0, stream>>>(nullptr, actA, wp3b, b3b, actB);
  maxpool2_bf16<<<(NB * 128 * 8 * 8 + 255) / 256, 256, 0, stream>>>(actB, actA, 128, 16, 16, NB * 128 * 8 * 8);
  conv3x3_wmma<128, 128, 8, 8, false><<<NB * 8 * 8 / 16, dim3(32, 8), 0, stream>>>(nullptr, actA, wp4a, b4a, actB);
  conv3x3_wmma<128, 128, 8, 8, false><<<NB * 8 * 8 / 16, dim3(32, 8), 0, stream>>>(nullptr, actB, wp4b, b4b, actA);

  // ---- FC head: actA viewed as (64, 8192) bf16 ----
  fc1_wmma<<<dim3(4, 64), 32, 0, stream>>>(actA, wp5, b5, fF);
  fc2_softmax<<<2, 32, 0, stream>>>(fF, w6, b6, hWs, outH);

  // ---- DLT solve + warp ----
  dlt_solve<<<2, 32, 0, stream>>>(hWs, CA, Hinv);
  warp_persp<<<(NB * IMG * IMG + 255) / 256, 256, 0, stream>>>(xa, Hinv, outW);
}
